// ArcFace_14474039788117
// MI455X (gfx1250) — matrix operand order repaired and checked by reading, not compiled, not run on hardware
//
#include <hip/hip_runtime.h>
#include <hip/hip_bf16.h>
#include <hip/hip_fp16.h>

typedef __attribute__((ext_vector_type(16))) _Float16 v16h;
typedef __attribute__((ext_vector_type(8)))  _Float16 v8h;
typedef __attribute__((ext_vector_type(8)))  float    v8f;

#define S_SCALE 30.0f
#define COS_M_  0.8775825618903728f
#define SIN_M_  0.479425538604203f
#define TH_    (-0.8775825618903728f)
#define MM_     0.2397127693021015f
#define C_CLASSES 100000

// Workspace layout:
//   [0, 512KB)        : A = normalized emb, f16, row-major [512][512]
//   [512KB, +2KB)     : rowsum f32[512]  (sum of exp(logit - S) per row)
//   [512KB+2KB, +2KB) : loglab f32[512]  (S * phi at the label column)

// ---------------------------------------------------------------------------
// Kernel 1: normalize embeddings -> f16 A buffer; zero rowsum accumulators.
// One wave per row (64 blocks * 8 waves = 512 rows).
// ---------------------------------------------------------------------------
__global__ __launch_bounds__(256) void arcface_prep(const float* __restrict__ img,
                                                    const float* __restrict__ prof,
                                                    _Float16* __restrict__ Abuf,
                                                    float* __restrict__ rowsum)
{
    const int tid  = threadIdx.x;
    const int wave = tid >> 5;
    const int lane = tid & 31;
    const int grow = blockIdx.x * 8 + wave;                    // 0..511

    const float* src = (grow < 256) ? (img  + (size_t)grow * 512)
                                    : (prof + (size_t)(grow - 256) * 512);
    float4 v[4];
    float  ss = 0.0f;
#pragma unroll
    for (int i = 0; i < 4; ++i) {
        v[i] = *(const float4*)(src + i * 128 + lane * 4);
        ss += v[i].x * v[i].x + v[i].y * v[i].y + v[i].z * v[i].z + v[i].w * v[i].w;
    }
#pragma unroll
    for (int m = 1; m < 32; m <<= 1) ss += __shfl_xor(ss, m, 32);
    const float inv = rsqrtf(ss);

    _Float16* dst = Abuf + (size_t)grow * 512;
#pragma unroll
    for (int i = 0; i < 4; ++i) {
        union { _Float16 h[4]; uint2 u; } pk;
        pk.h[0] = (_Float16)(v[i].x * inv);
        pk.h[1] = (_Float16)(v[i].y * inv);
        pk.h[2] = (_Float16)(v[i].z * inv);
        pk.h[3] = (_Float16)(v[i].w * inv);
        *(uint2*)(dst + i * 128 + lane * 4) = pk.u;
    }

    if (blockIdx.x == 0) {          // zero the 512 softmax accumulators
        rowsum[tid]       = 0.0f;
        rowsum[tid + 256] = 0.0f;
    }
}

// ---------------------------------------------------------------------------
// Kernel 2: fused weight-normalize + GEMM (WMMA f16->f32) + ArcFace margin +
// streaming softmax partial (fixed max = S).
// One block per 64 weight rows (last block covers only 32 valid rows).
// Block tile: all 512 emb rows x 64 cols. 8 waves: wave w -> rows [64w,64w+64),
// 4 row-tiles x 4 col-tiles of v_wmma_f32_16x16x32_f16; every A fragment is
// reused 4x, halving L2 traffic for the A matrix vs a 32-col tile.
// Epilogue validity is a per-element (c < C) predicate -> accumulators keep
// static indices and stay in registers (no scratch spills).
// ---------------------------------------------------------------------------
__global__ __launch_bounds__(256) void arcface_main(const float* __restrict__ weight,
                                                    const int*   __restrict__ label,
                                                    const _Float16* __restrict__ Abuf,
                                                    float* __restrict__ rowsum,
                                                    float* __restrict__ loglab)
{
    __shared__ _Float16 smW[64 * 512];   // normalized f16 weight tile, 64 KB
    __shared__ float    smN[64];         // per-row 1/||w||

    const int tid  = threadIdx.x;
    const int c0   = blockIdx.x * 64;
    const float* wt = weight + (size_t)c0 * 512;

    const int wv   = tid >> 5;           // wave 0..7
    const int ln   = tid & 31;
    const int mrow = ln & 15;
    const int hsel = ln >> 4;            // half-wave select

    // Prefetch this wave's A rows (gfx1250 global_prefetch_b8) while we
    // normalize the weight tile.
#pragma unroll
    for (int t = 0; t < 4; ++t)
        __builtin_prefetch(Abuf + (size_t)(wv * 64 + t * 16 + mrow) * 512, 0, 3);

    if (tid < 64) smN[tid] = 0.0f;
    __syncthreads();

    // Valid weight rows in this block's tile (last block: 32, else 64).
    const int nIter = (c0 + 32 < C_CLASSES) ? 32 : 16;   // float4 iters (256/iter)

    // Pass 1: per-row sum of squares (coalesced b128 reads; all 32 lanes of a
    // wave land in the same row each iteration -> wave shuffle reduce + one
    // LDS float atomic per wave per iteration).
    for (int i = 0; i < nIter; ++i) {
        const int q = i * 256 + tid;                      // float4 index in tile
        float4 w4 = *(const float4*)(wt + (size_t)q * 4);
        float ss = w4.x * w4.x + w4.y * w4.y + w4.z * w4.z + w4.w * w4.w;
#pragma unroll
        for (int m = 1; m < 32; m <<= 1) ss += __shfl_xor(ss, m, 32);
        if (ln == 0) atomicAdd(&smN[q >> 7], ss);         // 128 float4 per row
    }
    __syncthreads();
    if (tid < 64) smN[tid] = rsqrtf(smN[tid]);
    __syncthreads();

    // Pass 2: re-read (L2/L0 hit), normalize, convert, store f16 tile to LDS.
    for (int i = 0; i < nIter; ++i) {
        const int q = i * 256 + tid;
        float4 w4 = *(const float4*)(wt + (size_t)q * 4);
        const float inv = smN[q >> 7];
        union { _Float16 h[4]; uint2 u; } pk;
        pk.h[0] = (_Float16)(w4.x * inv);
        pk.h[1] = (_Float16)(w4.y * inv);
        pk.h[2] = (_Float16)(w4.z * inv);
        pk.h[3] = (_Float16)(w4.w * inv);
        *(uint2*)(&smW[q * 4]) = pk.u;
    }
    __syncthreads();

    // ---- WMMA main loop over K = 512 in steps of 32 ----
    const int koffA = hsel * 8;      // A frag: halves 0-7 -> K+koffA, 8-15 -> K+16+koffA
    const int koffB = hsel * 8;      // B frag: halves 0-7 -> K+koffB, 8-15 -> K+16+koffB
    v8f acc[4][4] = {};              // [row-tile][col-tile]
    const _Float16* aBase = Abuf + (size_t)(wv * 64 + mrow) * 512 + koffA;

    for (int k0 = 0; k0 < 512; k0 += 32) {
        v16h bf[4];
#pragma unroll
        for (int n = 0; n < 4; ++n) {
            const _Float16* bp = &smW[(n * 16 + mrow) * 512 + k0 + koffB];
            union { v16h v; v8h h[2]; } ub;
            ub.h[0] = *(const v8h*)(bp);
            ub.h[1] = *(const v8h*)(bp + 16);
            bf[n] = ub.v;
        }
        v16h af[4];
#pragma unroll
        for (int t = 0; t < 4; ++t) {
            const _Float16* ap = aBase + (size_t)t * (16 * 512) + k0;
            union { v16h v; v8h h[2]; } ua;
            ua.h[0] = *(const v8h*)(ap);
            ua.h[1] = *(const v8h*)(ap + 16);
            af[t] = ua.v;
        }
#pragma unroll
        for (int t = 0; t < 4; ++t)
#pragma unroll
            for (int n = 0; n < 4; ++n)
                acc[t][n] = __builtin_amdgcn_wmma_f32_16x16x32_f16(
                    false, af[t], false, bf[n], (short)0, acc[t][n], false, false);
    }

    // ---- Epilogue: ArcFace margin on label column + streaming exp-sum ----
    // Fully unrolled (static acc indices); validity via per-element c < C.
#pragma unroll
    for (int t = 0; t < 4; ++t) {
#pragma unroll
        for (int r = 0; r < 8; ++r) {
            const int grow = wv * 64 + t * 16 + r + 8 * hsel;     // emb row
            const int lab  = label[grow & 255];
            float sum = 0.0f;
#pragma unroll
            for (int n = 0; n < 4; ++n) {
                const float x = acc[t][n][r];                     // cosine
                const int   c = c0 + n * 16 + mrow;
                if (c < C_CLASSES) {
                    float val = x;
                    if (c == lab) {
                        const float sine = sqrtf(fmaxf(1.0f - x * x, 0.0f));
                        const float phi  = x * COS_M_ - sine * SIN_M_;
                        val = (x > TH_) ? phi : (x - MM_);
                        loglab[grow] = S_SCALE * val;             // unique writer
                    }
                    sum += __expf(S_SCALE * val - S_SCALE);       // fixed max = S
                }
            }
#pragma unroll
            for (int m = 1; m < 16; m <<= 1) sum += __shfl_xor(sum, m, 32);
            if (mrow == 0) atomicAdd(&rowsum[grow], sum);
        }
    }
}

// ---------------------------------------------------------------------------
// Kernel 3: nll[n] = S + log(rowsum[n]) - loglab[n]; out = mean.
// ---------------------------------------------------------------------------
__global__ __launch_bounds__(256) void arcface_finish(const float* __restrict__ rowsum,
                                                      const float* __restrict__ loglab,
                                                      float* __restrict__ out)
{
    __shared__ float sh[8];
    const int tid = threadIdx.x;
    float v = 0.0f;
#pragma unroll
    for (int i = 0; i < 2; ++i) {
        const int n = tid + i * 256;
        v += S_SCALE + __logf(rowsum[n]) - loglab[n];
    }
#pragma unroll
    for (int m = 1; m < 32; m <<= 1) v += __shfl_xor(v, m, 32);
    if ((tid & 31) == 0) sh[tid >> 5] = v;
    __syncthreads();
    if (tid == 0) {
        float s = 0.0f;
#pragma unroll
        for (int i = 0; i < 8; ++i) s += sh[i];
        *out = s * (1.0f / 512.0f);
    }
}

extern "C" void kernel_launch(void* const* d_in, const int* in_sizes, int n_in,
                              void* d_out, int out_size, void* d_ws, size_t ws_size,
                              hipStream_t stream) {
    const float* img    = (const float*)d_in[0];
    const float* prof   = (const float*)d_in[1];
    const float* weight = (const float*)d_in[2];
    const int*   label  = (const int*)d_in[3];
    float*       out    = (float*)d_out;

    char* ws = (char*)d_ws;
    _Float16* Abuf   = (_Float16*)ws;                       // 512 KB
    float*    rowsum = (float*)(ws + 512 * 512 * 2);        // 2 KB
    float*    loglab = rowsum + 512;                        // 2 KB

    arcface_prep  <<<64,   256, 0, stream>>>(img, prof, Abuf, rowsum);
    arcface_main  <<<1563, 256, 0, stream>>>(weight, label, Abuf, rowsum, loglab);
    arcface_finish<<<1,    256, 0, stream>>>(rowsum, loglab, out);
}
